// Attention_22668837388560
// MI455X (gfx1250) — compile-verified, matrix operand order checked by reading, not run on hardware
//
#include <hip/hip_runtime.h>

typedef __bf16 v16bf __attribute__((ext_vector_type(16)));
typedef float  v8f   __attribute__((ext_vector_type(8)));
typedef int    v4i   __attribute__((ext_vector_type(4)));
typedef __attribute__((address_space(3))) v4i v4i_lds;

#define DIMD  128
#define KEYS  16
#define TILE  16
#define WAVES 2

__device__ __forceinline__ unsigned short f32_to_bf16_rne(float f) {
  unsigned u = __float_as_uint(f);
  unsigned r = u + 0x7FFFu + ((u >> 16) & 1u);
  return (unsigned short)(r >> 16);
}
__device__ __forceinline__ float bf16_bits_to_f32(unsigned short s) {
  return __uint_as_float(((unsigned)s) << 16);
}
__device__ __forceinline__ __bf16 bf16_from_bits(unsigned short s) {
  return __builtin_bit_cast(__bf16, s);
}
// A-operand (16x32 bf16) element j -> K index for this lane-half
__device__ __forceinline__ int a_kidx(int j, int half) {
  int v = j >> 1;
  return ((v & 4) ? 16 : 0) + half * 8 + (v & 3) * 2 + (j & 1);
}
// async global->LDS 128-bit copy (per lane), tracked by ASYNCcnt
__device__ __forceinline__ void async_ld_b128(const float4* g, float* l) {
  __builtin_amdgcn_global_load_async_to_lds_b128((v4i*)g, (v4i_lds*)l, 0, 0);
}

// ---------------------------------------------------------------------------
// Prep: M = scale * Wk^T Wq (scale folded) + pre-swizzled bf16 hi/lo B-operand
// tiles for both M (scores path) and Wv (output projection path).
// ws layout (bf16): [0,16K) M_hi | [16K,32K) M_lo | [32K,48K) Wv_hi | [48K,64K) Wv_lo
// tile idx = ((c*8 + ntile)*32 + lane)*16 + j ; B[k][n],
// k = c*32 + a_kidx(j,half), n = ntile*16 + lane%16
// ---------------------------------------------------------------------------
__global__ __launch_bounds__(256) void attn_prep(const float* __restrict__ Wq,
                                                 const float* __restrict__ Wk,
                                                 const float* __restrict__ Wv,
                                                 __bf16* __restrict__ ws) {
  const int idx = blockIdx.x * 256 + threadIdx.x;   // 0 .. 16383
  const int j    = idx & 15;
  const int lane = (idx >> 4) & 31;
  const int nt   = (idx >> 9) & 7;
  const int c    = idx >> 12;
  const int half = lane >> 4;
  const int k = c * 32 + a_kidx(j, half);
  const int n = nt * 16 + (lane & 15);
  const float scale = 0.08838834764831845f;  // 1/sqrt(128)

  // M[n][k] = sum_j Wk[j][n] * Wq[j][k]  (B element for U = S @ M^T)
  float m = 0.f;
  for (int jj = 0; jj < DIMD; ++jj)
    m = fmaf(Wk[jj * DIMD + n], Wq[jj * DIMD + k], m);
  m *= scale;
  unsigned short mh = f32_to_bf16_rne(m);
  ws[idx]         = bf16_from_bits(mh);
  ws[16384 + idx] = bf16_from_bits(f32_to_bf16_rne(m - bf16_bits_to_f32(mh)));

  // B element for Out = W @ Wv^T : B[k][n] = Wv[n][k]
  float wv = Wv[n * DIMD + k];
  unsigned short vh = f32_to_bf16_rne(wv);
  ws[32768 + idx] = bf16_from_bits(vh);
  ws[49152 + idx] = bf16_from_bits(f32_to_bf16_rne(wv - bf16_bits_to_f32(vh)));
}

// ---------------------------------------------------------------------------
// Main: per wave, one 16-row tile.
//  U = S @ M^T (bf16 hi/lo WMMA) ; per-row: async-stream tgt row to LDS,
//  scores_i = tgt_i . u ; softmax ; w = sum attn_i tgt_i ; Out = W @ Wv^T
// ---------------------------------------------------------------------------
__global__ __launch_bounds__(64) void attn_main(const float* __restrict__ src,
                                                const float* __restrict__ tgt,
                                                const __bf16* __restrict__ MBh,
                                                const __bf16* __restrict__ MBl,
                                                const __bf16* __restrict__ VBh,
                                                const __bf16* __restrict__ VBl,
                                                float* __restrict__ out) {
  __shared__ float lds_S[WAVES][TILE][DIMD];        // 16 KB: S tile, reused as W tile
  __shared__ float lds_U[WAVES][TILE][DIMD];        // 16 KB: U tile
  __shared__ float lds_T[WAVES][2][KEYS][DIMD];     // 32 KB: tgt row double-buffer

  const int lane = threadIdx.x & 31;
  const int wave = threadIdx.x >> 5;
  const int m    = lane & 15;
  const int half = lane >> 4;
  const long long r0 = ((long long)blockIdx.x * WAVES + wave) * TILE;

  const float4* tgt4 = (const float4*)tgt;
  const size_t tbase = (size_t)r0 * (KEYS * DIMD / 4);  // float4 units

  // ---- issue async global->LDS stream of tgt row 0 (ASYNCcnt) ----
  {
    const float4* gp = tgt4 + tbase + lane;
    float* lp = &lds_T[wave][0][0][lane * 4];
    #pragma unroll
    for (int i = 0; i < KEYS; ++i)
      async_ld_b128(gp + i * (DIMD / 4), lp + i * DIMD);
  }

  // ---- stage S tile (16 x 128 f32) via B128 loads ----
  const float4* src4 = (const float4*)src;
  #pragma unroll
  for (int row = 0; row < TILE; ++row) {
    float4 v = src4[(size_t)(r0 + row) * (DIMD / 4) + lane];
    *(float4*)&lds_S[wave][row][lane * 4] = v;
  }
  __syncthreads();

  // ---- gather A operand (bf16 hi/lo split) from LDS ----
  v16bf a_hi[4], a_lo[4];
  #pragma unroll
  for (int c = 0; c < 4; ++c) {
    #pragma unroll
    for (int j = 0; j < 16; ++j) {
      float f = lds_S[wave][m][c * 32 + a_kidx(j, half)];
      unsigned short h = f32_to_bf16_rne(f);
      a_hi[c][j] = bf16_from_bits(h);
      a_lo[c][j] = bf16_from_bits(f32_to_bf16_rne(f - bf16_bits_to_f32(h)));
    }
  }

  // ---- U = S @ M^T via 3-product bf16 WMMA (hi*hi + hi*lo + lo*hi) ----
  const v16bf* MBh16 = (const v16bf*)MBh;
  const v16bf* MBl16 = (const v16bf*)MBl;
  #pragma unroll
  for (int nt = 0; nt < 8; ++nt) {
    v8f acc = {};
    #pragma unroll
    for (int c = 0; c < 4; ++c) {
      v16bf bh = MBh16[(c * 8 + nt) * 32 + lane];
      v16bf bl = MBl16[(c * 8 + nt) * 32 + lane];
      acc = __builtin_amdgcn_wmma_f32_16x16x32_bf16(false, a_hi[c], false, bh, (short)0, acc, false, false);
      acc = __builtin_amdgcn_wmma_f32_16x16x32_bf16(false, a_hi[c], false, bl, (short)0, acc, false, false);
      acc = __builtin_amdgcn_wmma_f32_16x16x32_bf16(false, a_lo[c], false, bh, (short)0, acc, false, false);
    }
    #pragma unroll
    for (int v = 0; v < 8; ++v)
      lds_U[wave][half * 8 + v][nt * 16 + m] = acc[v];
  }
  __syncthreads();

  // ---- per-row pipeline: async-load row r+1 while computing row r ----
  #pragma unroll 1
  for (int r = 0; r < TILE; ++r) {
    if (r + 1 < TILE) {
      // issue next row's async stream into the other buffer
      const float4* gp = tgt4 + tbase + (size_t)(r + 1) * (KEYS * DIMD / 4) + lane;
      float* lp = &lds_T[wave][(r + 1) & 1][0][lane * 4];
      #pragma unroll
      for (int i = 0; i < KEYS; ++i)
        async_ld_b128(gp + i * (DIMD / 4), lp + i * DIMD);
      // warm L2 two rows ahead
      if (r + 2 < TILE)
        __builtin_prefetch(tgt4 + tbase + (size_t)(r + 2) * (KEYS * DIMD / 4) + lane, 0, 1);
      __builtin_amdgcn_s_wait_asynccnt(16);   // row r resident (in-order completion)
    } else {
      __builtin_amdgcn_s_wait_asynccnt(0);
    }

    float4 td[KEYS];
    #pragma unroll
    for (int i = 0; i < KEYS; ++i)
      td[i] = *(const float4*)&lds_T[wave][r & 1][i][lane * 4];

    float4 u4 = *(const float4*)&lds_U[wave][r][lane * 4];
    float sc[KEYS];
    #pragma unroll
    for (int i = 0; i < KEYS; ++i) {
      float p = td[i].x * u4.x + td[i].y * u4.y + td[i].z * u4.z + td[i].w * u4.w;
      p += __shfl_xor(p, 16, 32);
      p += __shfl_xor(p, 8, 32);
      p += __shfl_xor(p, 4, 32);
      p += __shfl_xor(p, 2, 32);
      p += __shfl_xor(p, 1, 32);
      sc[i] = p;  // replicated across lanes; scale folded into M
    }
    float mx = sc[0];
    #pragma unroll
    for (int i = 1; i < KEYS; ++i) mx = fmaxf(mx, sc[i]);
    float ssum = 0.f;
    #pragma unroll
    for (int i = 0; i < KEYS; ++i) { sc[i] = __expf(sc[i] - mx); ssum += sc[i]; }
    const float inv = __frcp_rn(ssum);
    float4 wacc = {0.f, 0.f, 0.f, 0.f};
    #pragma unroll
    for (int i = 0; i < KEYS; ++i) {
      float a = sc[i] * inv;
      wacc.x = fmaf(a, td[i].x, wacc.x);
      wacc.y = fmaf(a, td[i].y, wacc.y);
      wacc.z = fmaf(a, td[i].z, wacc.z);
      wacc.w = fmaf(a, td[i].w, wacc.w);
    }
    *(float4*)&lds_S[wave][r][lane * 4] = wacc;  // reuse S tile as W tile
  }
  __syncthreads();

  // ---- Out = W @ Wv^T via 3-product bf16 WMMA ----
  #pragma unroll
  for (int c = 0; c < 4; ++c) {
    #pragma unroll
    for (int j = 0; j < 16; ++j) {
      float f = lds_S[wave][m][c * 32 + a_kidx(j, half)];
      unsigned short h = f32_to_bf16_rne(f);
      a_hi[c][j] = bf16_from_bits(h);
      a_lo[c][j] = bf16_from_bits(f32_to_bf16_rne(f - bf16_bits_to_f32(h)));
    }
  }
  const v16bf* VBh16 = (const v16bf*)VBh;
  const v16bf* VBl16 = (const v16bf*)VBl;
  #pragma unroll
  for (int nt = 0; nt < 8; ++nt) {
    v8f acc = {};
    #pragma unroll
    for (int c = 0; c < 4; ++c) {
      v16bf bh = VBh16[(c * 8 + nt) * 32 + lane];
      v16bf bl = VBl16[(c * 8 + nt) * 32 + lane];
      acc = __builtin_amdgcn_wmma_f32_16x16x32_bf16(false, a_hi[c], false, bh, (short)0, acc, false, false);
      acc = __builtin_amdgcn_wmma_f32_16x16x32_bf16(false, a_hi[c], false, bl, (short)0, acc, false, false);
      acc = __builtin_amdgcn_wmma_f32_16x16x32_bf16(false, a_lo[c], false, bh, (short)0, acc, false, false);
    }
    #pragma unroll
    for (int v = 0; v < 8; ++v)
      out[(size_t)(r0 + half * 8 + v) * DIMD + nt * 16 + m] = acc[v];
  }
}

extern "C" void kernel_launch(void* const* d_in, const int* in_sizes, int n_in,
                              void* d_out, int out_size, void* d_ws, size_t ws_size,
                              hipStream_t stream) {
  (void)n_in; (void)out_size; (void)ws_size;
  const float* src = (const float*)d_in[0];
  const float* tgt = (const float*)d_in[1];
  const float* Wq  = (const float*)d_in[2];
  const float* Wk  = (const float*)d_in[3];
  const float* Wv  = (const float*)d_in[4];
  __bf16* ws = (__bf16*)d_ws;

  // Stage 0: precompute M = scale*Wk^T@Wq and swizzle M/Wv into WMMA B layout.
  attn_prep<<<dim3(64), dim3(256), 0, stream>>>(Wq, Wk, Wv, ws);

  const __bf16* MBh = ws;
  const __bf16* MBl = ws + 16384;
  const __bf16* VBh = ws + 32768;
  const __bf16* VBl = ws + 49152;

  const int rows   = in_sizes[0] / DIMD;           // b*n = 32768
  const int blocks = rows / (TILE * WAVES);        // 1024 (exact for this shape)
  attn_main<<<dim3(blocks), dim3(32 * WAVES), 0, stream>>>(
      src, tgt, MBh, MBl, VBh, VBl, (float*)d_out);
}
